// Eyettention_27668179321129
// MI455X (gfx1250) — matrix operand order repaired and checked by reading, not compiled, not run on hardware
//
#include <hip/hip_runtime.h>

namespace {
constexpr int kB   = 128;   // batch
constexpr int kLSN = 256;   // subword seq len (encoder side)
constexpr int kSN  = 128;   // word seq len (encoder)
constexpr int kSP  = 128;   // decoder steps
constexpr int kD   = 768;   // embedding dim
constexpr int kH   = 512;   // decoder hidden / encoder concat
constexpr int kHH  = 256;   // encoder per-direction hidden
constexpr int kDinPad = 800; // 769 padded up to a multiple of 32 for WMMA K
constexpr float kNEG = -1.0e9f;
}

// ---------------- WMMA bf16 plumbing (CDNA5, wave32) ----------------
typedef __attribute__((ext_vector_type(16))) __bf16 v16bf;
typedef __attribute__((ext_vector_type(8)))  float  v8f;
typedef __attribute__((ext_vector_type(4)))  unsigned int v4ui;
typedef __attribute__((ext_vector_type(8)))  int v8i_;
typedef __attribute__((ext_vector_type(4)))  int v4i_;

union Frag16 { uint4 q[2]; v16bf v; };

__device__ __forceinline__ unsigned short f2bf(float f) {
  unsigned u = __float_as_uint(f);
  u += 0x7FFFu + ((u >> 16) & 1u);      // round-to-nearest-even
  return (unsigned short)(u >> 16);
}
__device__ __forceinline__ float bf2f(unsigned short s) {
  return __uint_as_float(((unsigned)s) << 16);
}
__device__ __forceinline__ float sigf(float x) { return 1.f / (1.f + __expf(-x)); }

// A fragment (16x32 bf16) of a row-major [16, stride] block starting at `base`, k-offset k0.
// ISA 7.12.2: lane l holds row (l&15); halves 0..7 = K k0+8*(l>>4)+i, halves 8..15 = K k0+16+8*(l>>4)+i.
__device__ __forceinline__ v16bf load_frag_A(const unsigned short* base, int stride, int k0, int lane) {
  const unsigned short* p = base + (size_t)(lane & 15) * stride + k0 + ((lane >> 4) << 3);
  Frag16 f;
  f.q[0] = *(const uint4*)p;
  f.q[1] = *(const uint4*)(p + 16);
  return f.v;
}
// B fragment (32x16 bf16) = W^T tile, sourced from row-major weight W[N, stride] rows n0..n0+15.
// Lane l holds W row n0+(l&15); halves 0..7 = K k0+8*(l>>4)+i, halves 8..15 = K k0+16+8*(l>>4)+i.
__device__ __forceinline__ v16bf load_frag_B(const unsigned short* base, int stride, int n0, int k0, int lane) {
  const unsigned short* p = base + (size_t)(n0 + (lane & 15)) * stride + k0 + ((lane >> 4) << 3);
  Frag16 f;
  f.q[0] = *(const uint4*)p;
  f.q[1] = *(const uint4*)(p + 16);
  return f.v;
}
__device__ __forceinline__ v8f wmma_bf16(v16bf a, v16bf b, v8f c) {
  return __builtin_amdgcn_wmma_f32_16x16x32_bf16(false, a, false, b, (short)0, c, false, false);
}

// LDS byte offset of a __shared__ object: generic LDS-aperture addresses carry the
// wave-relative LDS offset in addr[31:0] (ISA 10.2 aperture mapping).
__device__ __forceinline__ unsigned lds_addr_of(const void* p) {
  return (unsigned)(uintptr_t)p;
}

// ---------------- Tensor Data Mover: strided 2D bf16 tile -> LDS ----------------
// Builds a D# per CDNA5 ISA: group0 = {count=1, lds_addr,
// global_addr, type=2}; group1 = {data_size=2B, tensor described as exactly the
// tile (tile_d1 rows x tile_d0 elems) with row stride `row_stride` elements}.
__device__ __forceinline__ void tdm_load_2d_bf16(
    unsigned lds_byte_off, const unsigned short* gtile,
    unsigned tile_d0 /*elems/row*/, unsigned tile_d1 /*rows*/, unsigned row_stride /*elems*/) {
  unsigned long long ga = (unsigned long long)(uintptr_t)gtile;
  v4ui g0;
  g0.x = 1u;                                                 // count=1, user mode, no gather
  g0.y = lds_byte_off;                                       // lds_addr [63:32]
  g0.z = (unsigned)(ga & 0xFFFFFFFFu);                       // global_addr[31:0]
  g0.w = (unsigned)((ga >> 32) & 0x1FFFFFFu) | (2u << 30);   // global_addr[56:32] | type=2
  v8i_ g1;
  g1[0] = (int)(1u << 16);                                   // workgroup_mask=0, data_size=1 (2B)
  g1[1] = (int)((tile_d0 & 0xFFFFu) << 16);                  // tensor_dim0[15:0] (<<48)
  g1[2] = (int)(((tile_d0 >> 16) & 0xFFFFu) | ((tile_d1 & 0xFFFFu) << 16)); // dim0 hi | dim1 lo
  g1[3] = (int)(((tile_d1 >> 16) & 0xFFFFu) | ((tile_d0 & 0xFFFFu) << 16)); // dim1 hi | tile_dim0
  g1[4] = (int)(tile_d1 & 0xFFFFu);                          // tile_dim1 ; tile_dim2=0
  g1[5] = (int)row_stride;                                   // tensor_dim0_stride[31:0]
  g1[6] = 0;                                                 // stride hi / dim1_stride lo
  g1[7] = 0;
  v4i_ g2 = {0, 0, 0, 0};
  v4i_ g3 = {0, 0, 0, 0};
#if defined(__clang_major__) && (__clang_major__ >= 23)
  v8i_ g4 = {0, 0, 0, 0, 0, 0, 0, 0};
  __builtin_amdgcn_tensor_load_to_lds(g0, g1, g2, g3, g4, 0);
#else
  __builtin_amdgcn_tensor_load_to_lds(g0, g1, g2, g3, 0);
#endif
}

// ---------------- precision conversion kernels ----------------
__global__ void k_f32_to_bf16(const float* __restrict__ in, unsigned short* __restrict__ out, size_t n) {
  size_t i = (size_t)blockIdx.x * 256 + threadIdx.x;
  if (i < n) out[i] = f2bf(in[i]);
}
__global__ void k_cvt_pad(const float* __restrict__ in, unsigned short* __restrict__ out,
                          int rows, int incols, int outcols) {
  size_t i = (size_t)blockIdx.x * 256 + threadIdx.x;
  if (i >= (size_t)rows * outcols) return;
  int c = (int)(i % outcols); int r = (int)(i / outcols);
  out[i] = (c < incols) ? f2bf(in[(size_t)r * incols + c]) : (unsigned short)0;
}

// ---------------- subword->word pooling (encoder side) ----------------
__global__ __launch_bounds__(256) void pool_sn_kernel(
    const int* __restrict__ sn_emd, const int* __restrict__ wid,
    const float* __restrict__ word_emb,
    unsigned short* __restrict__ x0, int* __restrict__ mask)
{
  const int b = blockIdx.x, w = blockIdx.y, tid = threadIdx.x;
  __shared__ int s_tok[kLSN]; __shared__ int s_wid[kLSN];
  __shared__ float red[256];
  for (int l = tid; l < kLSN; l += 256) { s_tok[l] = sn_emd[b * kLSN + l]; s_wid[l] = wid[b * kLSN + l]; }
  __syncthreads();
  float acc[3] = {0.f, 0.f, 0.f};
  for (int l = 0; l < kLSN; ++l) {
    if (s_wid[l] == w) {   // uniform branch across block
      const float* e = word_emb + (size_t)s_tok[l] * kD;
      #pragma unroll
      for (int j = 0; j < 3; ++j) acc[j] += e[tid + 256 * j];
    }
  }
  const size_t base = ((size_t)w * kB + b) * kD;
  #pragma unroll
  for (int j = 0; j < 3; ++j) x0[base + tid + 256 * j] = f2bf(acc[j]);
  red[tid] = acc[0] + acc[1] + acc[2];
  __syncthreads();
  for (int o = 128; o > 0; o >>= 1) { if (tid < o) red[tid] += red[tid + o]; __syncthreads(); }
  if (tid == 0) mask[b * kSN + w] = (red[0] != 0.f) ? 1 : 0;
}

// ---------------- decoder input: pool + pos_emb + LayerNorm + fix_dur, padded to 800 ----------------
__global__ __launch_bounds__(256) void pool_sp_ln_kernel(
    const int* __restrict__ sp_emd, const int* __restrict__ wid_sp,
    const float* __restrict__ word_emb, const int* __restrict__ sp_pos,
    const float* __restrict__ pos_emb, const float* __restrict__ ln_g,
    const float* __restrict__ ln_b, const float* __restrict__ sp_fix_dur,
    unsigned short* __restrict__ dec_in /* (SP, B, 800) bf16 */)
{
  const int b = blockIdx.x, w = blockIdx.y, tid = threadIdx.x;
  __shared__ int s_tok[kSP]; __shared__ int s_wid[kSP];
  __shared__ float red[256];
  if (tid < kSP) { s_tok[tid] = sp_emd[b * (kSP + 1) + tid]; s_wid[tid] = wid_sp[b * (kSP + 1) + tid]; }
  __syncthreads();
  float x[3] = {0.f, 0.f, 0.f};
  for (int l = 0; l < kSP; ++l) {
    if (s_wid[l] == w) {
      const float* e = word_emb + (size_t)s_tok[l] * kD;
      #pragma unroll
      for (int j = 0; j < 3; ++j) x[j] += e[tid + 256 * j];
    }
  }
  {
    const float* pe = pos_emb + (size_t)sp_pos[b * kSP + w] * kD;
    #pragma unroll
    for (int j = 0; j < 3; ++j) x[j] += pe[tid + 256 * j];
  }
  red[tid] = x[0] + x[1] + x[2];
  __syncthreads();
  for (int o = 128; o > 0; o >>= 1) { if (tid < o) red[tid] += red[tid + o]; __syncthreads(); }
  const float mu = red[0] / (float)kD;
  __syncthreads();
  float sq = 0.f;
  #pragma unroll
  for (int j = 0; j < 3; ++j) { float d = x[j] - mu; sq += d * d; }
  red[tid] = sq;
  __syncthreads();
  for (int o = 128; o > 0; o >>= 1) { if (tid < o) red[tid] += red[tid + o]; __syncthreads(); }
  const float rstd = rsqrtf(red[0] / (float)kD + 1e-12f);
  const size_t base = ((size_t)w * kB + b) * kDinPad;
  #pragma unroll
  for (int j = 0; j < 3; ++j) {
    int d = tid + 256 * j;
    dec_in[base + d] = f2bf((x[j] - mu) * rstd * ln_g[d] + ln_b[d]);
  }
  if (tid == 0) dec_in[base + kD] = f2bf(sp_fix_dur[b * kSP + w]);
  for (int d = kD + 1 + tid; d < kDinPad; d += 256) dec_in[base + d] = 0;  // zero pad
}

// ---------------- WMMA GEMM with TDM-staged B tiles ----------------
// C[M,N] = A[M,K](bf16) @ W[N,K]^T(bf16) + bias. grid=(M/128, N/64), 256 thr = 8 waves.
// All 8 waves share one 64xK B-slab, so the 64x32 bf16 B tile is DMAed to LDS ONCE per
// block per k-chunk by the Tensor Data Mover (double-buffered, TENSORcnt-tracked),
// overlapping the DMA with WMMA on the previous chunk. A streams via global_load_b128.
__global__ __launch_bounds__(256) void gemm_bf16_bias(
    const unsigned short* __restrict__ A, const unsigned short* __restrict__ Bw,
    const float* __restrict__ bias, float* __restrict__ C, int N, int K)
{
  const int lane = threadIdx.x & 31, wave = threadIdx.x >> 5;
  const int m0 = blockIdx.x * 128 + wave * 16;
  const int n0 = blockIdx.y * 64;
  const unsigned short* Arow = A + (size_t)m0 * K;
  __shared__ __align__(16) unsigned short sB[2][64 * 32];   // 2 x 4KB B tiles
  v8f acc[4];
  #pragma unroll
  for (int j = 0; j < 4; ++j)
    #pragma unroll
    for (int r = 0; r < 8; ++r) acc[j][r] = 0.f;

  const int nchunks = K / 32;
  const unsigned short* Btile = Bw + (size_t)n0 * K;
  if (wave == 0) {   // prologue DMA: chunk 0
    tdm_load_2d_bf16(lds_addr_of(&sB[0][0]), Btile, 32u, 64u, (unsigned)K);
    __builtin_amdgcn_s_wait_tensorcnt(0);
  }
  __syncthreads();

  for (int c = 0; c < nchunks; ++c) {
    const int k = c * 32;
    if (wave == 0 && c + 1 < nchunks)   // async DMA of next chunk into the other buffer
      tdm_load_2d_bf16(lds_addr_of(&sB[(c + 1) & 1][0]), Btile + (k + 32), 32u, 64u, (unsigned)K);
    v16bf a = load_frag_A(Arow, K, k, lane);
    if (k + 32 < K)
      __builtin_prefetch(Arow + (size_t)(lane & 15) * K + k + 32, 0, 3);
    const unsigned short* bt = &sB[c & 1][0];
    #pragma unroll
    for (int j = 0; j < 4; ++j) {
      v16bf b = load_frag_B(bt, 32, j * 16, 0, lane);   // ds_load_b128 from staged tile
      acc[j] = wmma_bf16(a, b, acc[j]);
    }
    if (wave == 0 && c + 1 < nchunks) __builtin_amdgcn_s_wait_tensorcnt(0);
    __syncthreads();
  }

  const int grp = lane >> 4, col = lane & 15;
  #pragma unroll
  for (int j = 0; j < 4; ++j) {
    const int n = n0 + j * 16 + col;
    const float bb = bias ? bias[n] : 0.f;
    #pragma unroll
    for (int r = 0; r < 8; ++r)
      C[(size_t)(m0 + r + 8 * grp) * N + n] = acc[j][r] + bb;
  }
}

// ---------------- encoder recurrent: per (batch-group, direction), 128 sequential steps ----------------
// G holds x@Wih^T + b (dir-major, 2048 wide). gates += h_{t-1} @ Whh^T via WMMA; LSTM cell in
// accumulator-fragment layout; c resident in VGPRs for all 128 steps; h double-buffered in LDS.
__global__ __launch_bounds__(256) void enc_recurrent(
    const float* __restrict__ G, const unsigned short* __restrict__ Whh,
    unsigned short* __restrict__ xnext /* (SN, B, 512) bf16 */)
{
  const int bg = blockIdx.x;     // 0..7 -> batch rows [16bg, 16bg+16)
  const int dir = blockIdx.y;    // 0 fwd, 1 bwd
  const int tid = threadIdx.x;
  const int lane = tid & 31, wave = tid >> 5;
  const int grp = lane >> 4, col = lane & 15;
  __shared__ __align__(16) unsigned short hbuf[2][16 * kHH];
  for (int i = tid; i < 16 * kHH; i += 256) { hbuf[0][i] = 0; hbuf[1][i] = 0; }
  __syncthreads();
  const unsigned short* Wd = Whh + (size_t)dir * 1024 * kHH;
  v8f creg[2];
  #pragma unroll
  for (int n = 0; n < 2; ++n)
    #pragma unroll
    for (int r = 0; r < 8; ++r) creg[n][r] = 0.f;

  for (int tt = 0; tt < kSN; ++tt) {
    const int t = dir ? (kSN - 1 - tt) : tt;
    const unsigned short* hprev = hbuf[tt & 1];
    unsigned short* hnext = hbuf[(tt + 1) & 1];
    const float* Gb = G + ((size_t)t * kB + bg * 16) * 2048 + dir * 1024;
    #pragma unroll
    for (int nbi = 0; nbi < 2; ++nbi) {
      const int nb = wave + nbi * 8;   // 0..15 column tiles of 256
      v8f gate[4];                     // i,f,g,o quad -- all in accumulator layout
      #pragma unroll
      for (int g = 0; g < 4; ++g)
        #pragma unroll
        for (int r = 0; r < 8; ++r)
          gate[g][r] = Gb[(size_t)(r + 8 * grp) * 2048 + g * kHH + nb * 16 + col];
      for (int k = 0; k < kHH; k += 32) {
        v16bf a = load_frag_A(hprev, kHH, k, lane);   // ds_load_b128 from LDS
        #pragma unroll
        for (int g = 0; g < 4; ++g)
          gate[g] = wmma_bf16(a, load_frag_B(Wd, kHH, g * kHH + nb * 16, k, lane), gate[g]);
      }
      #pragma unroll
      for (int r = 0; r < 8; ++r) {
        const int m = r + 8 * grp;
        const int j = nb * 16 + col;
        const float cc = sigf(gate[1][r]) * creg[nbi][r] + sigf(gate[0][r]) * tanhf(gate[2][r]);
        creg[nbi][r] = cc;
        const unsigned short hb = f2bf(sigf(gate[3][r]) * tanhf(cc));
        hnext[m * kHH + j] = hb;
        xnext[((size_t)t * kB + bg * 16 + m) * kH + dir * kHH + j] = hb;
      }
    }
    __syncthreads();
  }
}

// ---------------- enc_out: (b, s, 513) f32 = [h_concat, sn_word_len] ----------------
__global__ void build_enc_out(const unsigned short* __restrict__ xf,
                              const float* __restrict__ sn_word_len,
                              float* __restrict__ enc_out)
{
  size_t i = (size_t)blockIdx.x * 256 + threadIdx.x;
  const size_t total = (size_t)kB * kSN * 513;
  if (i >= total) return;
  int c = (int)(i % 513); size_t bs = i / 513; int s = (int)(bs % kSN); int b = (int)(bs / kSN);
  enc_out[i] = (c < kH) ? bf2f(xf[((size_t)s * kB + b) * kH + c]) : sn_word_len[b * kSN + s];
}

// ---------------- decoder: 8 stacked LSTM layers + windowed attention, fully sequential in t ----------------
__global__ __launch_bounds__(256) void dec_recurrent(
    const float* __restrict__ G1,             // (SP*B, 2048) = dec_in @ Wih1^T + b0
    const unsigned short* __restrict__ Wih,   // (7, 2048, 512) bf16
    const unsigned short* __restrict__ Whh,   // (8, 2048, 512) bf16
    const float* __restrict__ dec_bias,       // (8, 2048) f32
    const unsigned short* __restrict__ attnW, // (513, 512) bf16
    const float* __restrict__ attn_bias,      // (513) f32
    const float* __restrict__ enc_out,        // (B, SN, 513) f32
    const int* __restrict__ mask,             // (B, SN)
    const int* __restrict__ sp_pos,           // (B, SP)
    unsigned short* __restrict__ hstate,      // (8bg, 8, 16, 512) bf16 scratch
    float* __restrict__ cstate,               // (8bg, 8, 16, 512) f32 scratch
    float* __restrict__ out)                  // (B, SP, 513) f32
{
  const int bg = blockIdx.x;   // 0..7 -> batch rows [16bg, 16bg+16)
  const int tid = threadIdx.x;
  const int lane = tid & 31, wave = tid >> 5;
  const int grp = lane >> 4, col = lane & 15;
  __shared__ __align__(16) unsigned short hping[2][16 * kH];  // 32 KB: inter-layer h relay
  __shared__ __align__(16) unsigned short qbuf[16 * 520];     // ~16.6 KB: attention query
  unsigned short* hst = hstate + (size_t)bg * 8 * 16 * kH;
  float* cst = cstate + (size_t)bg * 8 * 16 * kH;
  for (int i = tid; i < 8 * 16 * kH; i += 256) { hst[i] = 0; cst[i] = 0.f; }
  __syncthreads();
  __threadfence();

  for (int t = 0; t < kSP; ++t) {
    for (int layer = 0; layer < 8; ++layer) {
      const unsigned short* hin = hping[layer & 1];          // layer-1 output, current t
      unsigned short* hout = hping[(layer + 1) & 1];
      const unsigned short* Wh = Whh + (size_t)layer * 2048 * kH;
      const unsigned short* hrec = hst + (size_t)layer * 16 * kH;  // this layer's h at t-1
      float* crec = cst + (size_t)layer * 16 * kH;
      #pragma unroll 1
      for (int nbi = 0; nbi < 4; ++nbi) {
        const int nb = wave + nbi * 8;   // 0..31 column tiles of 512
        v8f gate[4];
        if (layer == 0) {                // input GEMM precomputed into G1 (incl. bias+fix_dur col)
          const float* Gb = G1 + ((size_t)t * kB + bg * 16) * 2048;
          #pragma unroll
          for (int g = 0; g < 4; ++g)
            #pragma unroll
            for (int r = 0; r < 8; ++r)
              gate[g][r] = Gb[(size_t)(r + 8 * grp) * 2048 + g * kH + nb * 16 + col];
        } else {
          const unsigned short* Wi = Wih + (size_t)(layer - 1) * 2048 * kH;
          #pragma unroll
          for (int g = 0; g < 4; ++g) {
            const float bb = dec_bias[layer * 2048 + g * kH + nb * 16 + col];
            #pragma unroll
            for (int r = 0; r < 8; ++r) gate[g][r] = bb;
          }
          for (int k = 0; k < kH; k += 32) {
            v16bf a = load_frag_A(hin, kH, k, lane);
            #pragma unroll
            for (int g = 0; g < 4; ++g)
              gate[g] = wmma_bf16(a, load_frag_B(Wi, kH, g * kH + nb * 16, k, lane), gate[g]);
          }
        }
        for (int k = 0; k < kH; k += 32) {
          v16bf a = load_frag_A(hrec, kH, k, lane);
          #pragma unroll
          for (int g = 0; g < 4; ++g)
            gate[g] = wmma_bf16(a, load_frag_B(Wh, kH, g * kH + nb * 16, k, lane), gate[g]);
        }
        #pragma unroll
        for (int r = 0; r < 8; ++r) {
          const int m = r + 8 * grp;
          const int j = nb * 16 + col;
          const float cc = sigf(gate[1][r]) * crec[m * kH + j] + sigf(gate[0][r]) * tanhf(gate[2][r]);
          crec[m * kH + j] = cc;
          hout[m * kH + j] = f2bf(sigf(gate[3][r]) * tanhf(cc));
        }
      }
      __syncthreads();  // all reads of hrec / hin complete
      unsigned short* hdst = hst + (size_t)layer * 16 * kH;   // persist h for t+1
      for (int i = tid; i < 16 * kH; i += 256) hdst[i] = hout[i];
    }
    // ---- attention (h7 lives in hping[0]) ----
    const unsigned short* h7 = hping[0];
    #pragma unroll 1
    for (int nbi = 0; nbi < 4; ++nbi) {   // q[0..511] = h7 @ attnW^T + b
      const int nb = wave + nbi * 8;
      v8f q = {};
      for (int k = 0; k < kH; k += 32) {
        v16bf a = load_frag_A(h7, kH, k, lane);
        q = wmma_bf16(a, load_frag_B(attnW, kH, nb * 16, k, lane), q);
      }
      #pragma unroll
      for (int r = 0; r < 8; ++r) {
        const int n = nb * 16 + col;
        qbuf[(r + 8 * grp) * 520 + n] = f2bf(q[r] + attn_bias[n]);
      }
    }
    if (tid < 16) {                       // q[512] (attn_W row 512)
      float s = attn_bias[512];
      const unsigned short* wrow = attnW + (size_t)512 * kH;
      for (int k = 0; k < kH; ++k) s += bf2f(h7[tid * kH + k]) * bf2f(wrow[k]);
      qbuf[tid * 520 + 512] = f2bf(s);
    }
    __syncthreads();
    {
      // Window softmax: outside [pos-1,pos+1] gets +NEG=-1e9 -> exp underflows to exactly 0 in f32,
      // so restricting to the <=3 window positions is numerically identical to the reference.
      const int m = tid >> 4, cpart = tid & 15;
      const int b = bg * 16 + m;
      const int pos = sp_pos[b * kSP + t];
      const int left = (pos - 1 > 0) ? pos - 1 : 0;
      const int right = (pos + 1 < kSN - 1) ? pos + 1 : kSN - 1;
      const int ns = right - left + 1;
      float sc[3];
      #pragma unroll
      for (int si = 0; si < 3; ++si) {
        float p = 0.f;
        if (si < ns) {
          const float* e = enc_out + ((size_t)b * kSN + left + si) * 513;
          for (int c = cpart; c < 513; c += 16) p += bf2f(qbuf[m * 520 + c]) * e[c];
        }
        #pragma unroll
        for (int off = 8; off > 0; off >>= 1) p += __shfl_xor(p, off, 32);  // 16-lane butterfly
        if (si < ns && mask[b * kSN + left + si] == 0) p += kNEG;
        sc[si] = p;
      }
      float mx = -3.0e38f;
      for (int si = 0; si < ns; ++si) mx = fmaxf(mx, sc[si]);
      float wv[3]; float den = 0.f;
      #pragma unroll
      for (int si = 0; si < 3; ++si) { wv[si] = (si < ns) ? __expf(sc[si] - mx) : 0.f; den += wv[si]; }
      const float inv = 1.f / den;
      #pragma unroll
      for (int si = 0; si < 3; ++si) {
        const float d = (float)(left + si - pos);
        wv[si] *= inv * __expf(-2.f * d * d);   // gaussian exp(-(s-pos)^2/0.5)
      }
      for (int c = cpart; c < 513; c += 16) {
        float acc = 0.f;
        for (int si = 0; si < ns; ++si)
          acc += wv[si] * enc_out[((size_t)b * kSN + left + si) * 513 + c];
        out[((size_t)b * kSP + t) * 513 + c] = acc;
      }
    }
    __threadfence();
    __syncthreads();
  }
}

// ---------------- host orchestration ----------------
extern "C" void kernel_launch(void* const* d_in, const int* in_sizes, int n_in,
                              void* d_out, int out_size, void* d_ws, size_t ws_size,
                              hipStream_t stream) {
  (void)in_sizes; (void)n_in; (void)out_size; (void)ws_size;
  const int*   sn_emd      = (const int*)  d_in[0];
  const int*   word_ids_sn = (const int*)  d_in[2];
  const float* sn_word_len = (const float*)d_in[3];
  const int*   sp_emd      = (const int*)  d_in[4];
  const int*   sp_pos      = (const int*)  d_in[5];
  const int*   word_ids_sp = (const int*)  d_in[6];
  const float* sp_fix_dur  = (const float*)d_in[7];
  const float* word_emb    = (const float*)d_in[8];
  const float* pos_emb     = (const float*)d_in[9];
  const float* ln_g        = (const float*)d_in[10];
  const float* ln_b        = (const float*)d_in[11];
  const float* enc_Wih0    = (const float*)d_in[12];
  const float* enc_Wih     = (const float*)d_in[13];
  const float* enc_Whh     = (const float*)d_in[14];
  const float* enc_b       = (const float*)d_in[15];
  const float* dec_Wih1    = (const float*)d_in[16];
  const float* dec_Wih     = (const float*)d_in[17];
  const float* dec_Whh     = (const float*)d_in[18];
  const float* dec_b       = (const float*)d_in[19];
  const float* attn_W      = (const float*)d_in[20];
  const float* attn_b      = (const float*)d_in[21];

  char* ws = (char*)d_ws;
  size_t off = 0;
  auto alloc = [&](size_t bytes) -> char* {
    off = (off + 255) & ~(size_t)255;
    char* p = ws + off; off += bytes; return p;
  };
  unsigned short* w_eWih0 = (unsigned short*)alloc((size_t)2 * 1024 * 768 * 2);
  unsigned short* w_eWih  = (unsigned short*)alloc((size_t)7 * 2 * 1024 * 512 * 2);
  unsigned short* w_eWhh  = (unsigned short*)alloc((size_t)8 * 2 * 1024 * 256 * 2);
  unsigned short* w_dWih1 = (unsigned short*)alloc((size_t)2048 * kDinPad * 2);
  unsigned short* w_dWih  = (unsigned short*)alloc((size_t)7 * 2048 * 512 * 2);
  unsigned short* w_dWhh  = (unsigned short*)alloc((size_t)8 * 2048 * 512 * 2);
  unsigned short* w_attn  = (unsigned short*)alloc((size_t)513 * 512 * 2);
  unsigned short* x0      = (unsigned short*)alloc((size_t)kSN * kB * 768 * 2);
  unsigned short* xb1     = (unsigned short*)alloc((size_t)kSN * kB * 512 * 2);
  unsigned short* xb2     = (unsigned short*)alloc((size_t)kSN * kB * 512 * 2);
  unsigned short* decin   = (unsigned short*)alloc((size_t)kSP * kB * kDinPad * 2);
  float* G       = (float*)alloc((size_t)16384 * 2048 * 4);      // shared gate-precompute buffer
  float* enc_out = (float*)alloc((size_t)kB * kSN * 513 * 4);
  int*   maskbuf = (int*)  alloc((size_t)kB * kSN * 4);
  unsigned short* hstate = (unsigned short*)alloc((size_t)8 * 8 * 16 * 512 * 2);
  float* cstate  = (float*)alloc((size_t)8 * 8 * 16 * 512 * 4);

  auto cvt = [&](const float* in, unsigned short* out_, size_t n) {
    k_f32_to_bf16<<<dim3((unsigned)((n + 255) / 256)), dim3(256), 0, stream>>>(in, out_, n);
  };
  cvt(enc_Wih0, w_eWih0, (size_t)2 * 1024 * 768);
  cvt(enc_Wih,  w_eWih,  (size_t)7 * 2 * 1024 * 512);
  cvt(enc_Whh,  w_eWhh,  (size_t)8 * 2 * 1024 * 256);
  cvt(dec_Wih,  w_dWih,  (size_t)7 * 2048 * 512);
  cvt(dec_Whh,  w_dWhh,  (size_t)8 * 2048 * 512);
  cvt(attn_W,   w_attn,  (size_t)513 * 512);
  {
    size_t n = (size_t)2048 * kDinPad;
    k_cvt_pad<<<dim3((unsigned)((n + 255) / 256)), 256, 0, stream>>>(dec_Wih1, w_dWih1, 2048, 769, kDinPad);
  }

  pool_sn_kernel<<<dim3(kB, kSN), 256, 0, stream>>>(sn_emd, word_ids_sn, word_emb, x0, maskbuf);
  pool_sp_ln_kernel<<<dim3(kB, kSP), 256, 0, stream>>>(sp_emd, word_ids_sp, word_emb, sp_pos,
                                                       pos_emb, ln_g, ln_b, sp_fix_dur, decin);

  // 8-layer BiLSTM encoder: big parallel WMMA GEMM (x @ Wih^T + b, both dirs as N=2048), then recurrence.
  const unsigned short* xin = x0;
  unsigned short* xout = xb1;
  for (int l = 0; l < 8; ++l) {
    const int K = (l == 0) ? 768 : 512;
    const unsigned short* W = (l == 0) ? w_eWih0 : (w_eWih + (size_t)(l - 1) * 2 * 1024 * 512);
    gemm_bf16_bias<<<dim3(128, 32), 256, 0, stream>>>(xin, W, enc_b + (size_t)l * 2048, G, 2048, K);
    enc_recurrent<<<dim3(8, 2), 256, 0, stream>>>(G, w_eWhh + (size_t)l * 2 * 1024 * 256, xout);
    xin = xout;
    xout = (xout == xb1) ? xb2 : xb1;
  }
  {
    size_t n = (size_t)kB * kSN * 513;
    build_enc_out<<<dim3((unsigned)((n + 255) / 256)), 256, 0, stream>>>(xin, sn_word_len, enc_out);
  }

  // Decoder layer-0 input GEMM for all timesteps (K padded 769->800), then sequential decoder+attention.
  gemm_bf16_bias<<<dim3(128, 32), 256, 0, stream>>>(decin, w_dWih1, dec_b, G, 2048, kDinPad);
  dec_recurrent<<<dim3(8), 256, 0, stream>>>(G, w_dWih, w_dWhh, dec_b, w_attn, attn_b, enc_out,
                                             maskbuf, sp_pos, hstate, cstate, (float*)d_out);
}
